// WideHyperBasicBlock_20126216749252
// MI455X (gfx1250) — matrix operand order repaired and checked by reading, not compiled, not run on hardware
//
#include <hip/hip_runtime.h>

// ---------------------------------------------------------------------------
// WideHyperBasicBlock for MI455X (gfx1250, wave32, WMMA bf16 16x16x32).
// HD==1 => tw[b] = h_b * W + Bias. Precompute bf16 scaled weights with the K
// dimension permuted tap-major (k' = tap*256 + ci) so every 64-wide K chunk of
// the implicit GEMM lives inside one filter tap (uniform im2col addressing).
//   GEMM per sample: M = Cout = 256, N = H*W = 4096, K = 2304.
// ---------------------------------------------------------------------------

typedef unsigned short ushort_t;
typedef __attribute__((ext_vector_type(16))) __bf16 v16bf;
typedef __attribute__((ext_vector_type(8)))  __bf16 v8bf;
typedef __attribute__((ext_vector_type(8)))  float  v8f;
typedef int v4i_t __attribute__((ext_vector_type(4)));
typedef __attribute__((address_space(1))) v4i_t* as1_v4i_p;   // global int4*
typedef __attribute__((address_space(3))) v4i_t* as3_v4i_p;   // LDS int4*

#define BN_EPS 1e-5f
#define HW    4096
#define KDIM  2304
#define KCH   64            // K chunk per LDS stage (2 WMMA sub-steps)

#if defined(__has_builtin)
#if __has_builtin(__builtin_amdgcn_global_load_async_to_lds_b128)
#define HAVE_ASYNC_COPY 1
#endif
#endif
#ifndef HAVE_ASYNC_COPY
#define HAVE_ASYNC_COPY 0
#endif

__device__ __forceinline__ ushort_t f2bf(float f) {
  union { float f; unsigned u; } c; c.f = f;
  unsigned r = c.u + 0x7FFFu + ((c.u >> 16) & 1u);   // round-to-nearest-even
  return (ushort_t)(r >> 16);
}
__device__ __forceinline__ unsigned packbf2(float a, float b) {
  return (unsigned)f2bf(a) | ((unsigned)f2bf(b) << 16);
}

#if HAVE_ASYNC_COPY
template<int OFF>
__device__ __forceinline__ void async_cp16(const ushort_t* g, ushort_t* l) {
  // AS1 pointer: full 64-bit VA.  AS3 pointer: low 32 bits = LDS offset
  // (ISA aperture rule).
  as1_v4i_p gp = (as1_v4i_p)(unsigned long long)(uintptr_t)g;
  as3_v4i_p lp = (as3_v4i_p)(unsigned)(uintptr_t)l;
  __builtin_amdgcn_global_load_async_to_lds_b128(gp, lp, OFF, 0);
}
#endif
__device__ __forceinline__ void wait_async0() {
#if defined(__has_builtin) && __has_builtin(__builtin_amdgcn_s_wait_asynccnt)
  __builtin_amdgcn_s_wait_asynccnt(0);
#else
  asm volatile("s_wait_asynccnt 0x0" ::: "memory");
#endif
}

// ---------------------------------------------------------------------------
// Kernel 1: A1 = bf16(relu(bn1(x)))
// ---------------------------------------------------------------------------
__global__ __launch_bounds__(256)
void prep_bn_relu(const float* __restrict__ x,
                  const float* __restrict__ gamma, const float* __restrict__ beta,
                  const float* __restrict__ mean,  const float* __restrict__ var,
                  ushort_t* __restrict__ a1) {
  int i = blockIdx.x * 256 + threadIdx.x;       // float4 index
  int c = ((i << 2) >> 12) & 255;
  float inv  = gamma[c] * rsqrtf(var[c] + BN_EPS);
  float addv = fmaf(-mean[c], inv, beta[c]);
  float4 v = ((const float4*)x)[i];
  float y0 = fmaxf(fmaf(v.x, inv, addv), 0.f);
  float y1 = fmaxf(fmaf(v.y, inv, addv), 0.f);
  float y2 = fmaxf(fmaf(v.z, inv, addv), 0.f);
  float y3 = fmaxf(fmaf(v.w, inv, addv), 0.f);
  uint2 o; o.x = packbf2(y0, y1); o.y = packbf2(y2, y3);
  ((uint2*)a1)[i] = o;
}

// ---------------------------------------------------------------------------
// Kernel 2: Wb[b][o][tap*256+ci] = bf16(h_b * w[o][ci][tap] + bias[o][ci][tap])
// One uint (2 ci) per thread; writes coalesced, reads served from L2 (2.3 MB).
// ---------------------------------------------------------------------------
__global__ __launch_bounds__(256)
void prep_weights(const float* __restrict__ w, const float* __restrict__ bias,
                  const float* __restrict__ h_in, unsigned* __restrict__ wbout) {
  unsigned tid = blockIdx.x * 256 + threadIdx.x;  // < 32*256*9*128 = 9,437,184
  int ci       = (tid & 127) << 1;
  unsigned t2  = tid >> 7;                        // (b*256 + o)*9 + tap
  int tap      = t2 % 9u;
  unsigned t3  = t2 / 9u;                         // b*256 + o
  int o        = t3 & 255;
  int bb       = t3 >> 8;
  float h = 0.5f + h_in[bb] * (1.0f / 64.0f);
  size_t i0 = (size_t)o * KDIM + (size_t)ci * 9 + tap;
  float v0 = fmaf(h, w[i0],     bias[i0]);
  float v1 = fmaf(h, w[i0 + 9], bias[i0 + 9]);
  wbout[tid] = packbf2(v0, v1);
}

// ---------------------------------------------------------------------------
// Implicit-GEMM conv. 128x128 tile/block, 8 waves (4x2 of 32x64), K chunks of
// 64 (one tap), double-buffered LDS, async global->LDS for the weight tile.
// STAGE 1: bn2+relu -> bf16 actout.  STAGE 2: + residual -> f32 out.
// ---------------------------------------------------------------------------
template<int STAGE>
__global__ __launch_bounds__(256)
void hyperconv_wmma(const ushort_t* __restrict__ act,   // [32,256,64,64] bf16
                    const ushort_t* __restrict__ wb,    // [32,256,2304] bf16 tap-major K
                    const float* __restrict__ g2, const float* __restrict__ be2,
                    const float* __restrict__ mn2, const float* __restrict__ vr2,
                    const float* __restrict__ xres,     // residual (STAGE 2)
                    float* __restrict__ out,            // f32 out   (STAGE 2)
                    ushort_t* __restrict__ actout) {    // bf16 out  (STAGE 1)
  __shared__ alignas(16) ushort_t ldsA[2][128 * KCH];   // [m][k'] 128B rows, 2x16KB
  __shared__ alignas(16) ushort_t ldsB[2][128 * KCH];   // [n][k'] 128B rows, 2x16KB

  const int b    = blockIdx.z;
  const int m0   = blockIdx.y * 128;
  const int n0   = blockIdx.x * 128;
  const int t    = threadIdx.x;
  const int lane = t & 31, wid = t >> 5;
  const int half = lane >> 4, lm = lane & 15;
  const int wm   = wid & 3,  wn  = wid >> 2;            // wave tile 32x64

  const ushort_t* actb = act + ((size_t)b << 20);       // b*256*4096
  const ushort_t* wbb  = wb  + (size_t)b * 256 * KDIM;

  const int lrow = t >> 1;                              // tile row this thread fills
  const int lseg = (t & 1) * 32;                        // K half of the 64-wide row
  const int p  = n0 + lrow;                             // this thread's pixel (B tile)
  const int py = p >> 6, px = p & 63;

  const ushort_t* aSrc = wbb + (size_t)(m0 + lrow) * KDIM + lseg;

  v8f zero = {};
  v8f acc[2][4];
  #pragma unroll
  for (int i = 0; i < 2; ++i)
    #pragma unroll
    for (int j = 0; j < 4; ++j) acc[i][j] = zero;

#if !HAVE_ASYNC_COPY
  uint4 aregs[4];
#endif
  alignas(16) ushort_t valsB[32];

  auto issueA = [&](int kk, int buf) {
    const ushort_t* g = aSrc + kk;
    ushort_t* l = &ldsA[buf][lrow * KCH + lseg];
#if HAVE_ASYNC_COPY
    async_cp16<0>(g, l); async_cp16<16>(g, l);
    async_cp16<32>(g, l); async_cp16<48>(g, l);
#else
    const uint4* gs = (const uint4*)g;
    aregs[0] = gs[0]; aregs[1] = gs[1]; aregs[2] = gs[2]; aregs[3] = gs[3];
    (void)l;
#endif
  };
  auto commitA = [&](int buf) {
#if HAVE_ASYNC_COPY
    wait_async0();
    (void)buf;
#else
    uint4* d = (uint4*)&ldsA[buf][lrow * KCH + lseg];
    d[0] = aregs[0]; d[1] = aregs[1]; d[2] = aregs[2]; d[3] = aregs[3];
#endif
  };
  auto gatherB = [&](int kk) {
    // k' = tap*256 + ci : whole 64-chunk has one tap -> uniform (sy,sx)
    int tap = kk >> 8;
    int kh = tap / 3, kw = tap - kh * 3;
    int sy = py + kh - 1, sx = px + kw - 1;
    int ci0 = (kk & 255) + lseg;
    if ((unsigned)sy < 64u && (unsigned)sx < 64u) {
      const ushort_t* src = actb + ((size_t)ci0 << 12) + (sy << 6) + sx;
      #pragma unroll
      for (int i = 0; i < 32; ++i) valsB[i] = src[(size_t)i << 12];
    } else {
      #pragma unroll
      for (int i = 0; i < 32; ++i) valsB[i] = 0;
    }
  };
  auto commitB = [&](int buf) {
    uint4* d = (uint4*)&ldsB[buf][lrow * KCH + lseg];
    const uint4* s = (const uint4*)valsB;
    d[0] = s[0]; d[1] = s[1]; d[2] = s[2]; d[3] = s[3];
  };
  auto compute = [&](int buf) {
    #pragma unroll
    for (int s = 0; s < 2; ++s) {                       // two 16x16x32 sub-steps
      v16bf afrag[2], bfrag[4];
      #pragma unroll
      for (int i = 0; i < 2; ++i) {
        int row = wm * 32 + i * 16 + lm;
        const ushort_t* base = &ldsA[buf][row * KCH + s * 32];
        // lane 0-15: K 0-7 & 16-23 ; lane 16-31: K 8-15 & 24-31
        v8bf lo = *(const v8bf*)(base + half * 8);
        v8bf hi = *(const v8bf*)(base + half * 8 + 16);
        afrag[i] = __builtin_shufflevector(lo, hi,
                     0,1,2,3,4,5,6,7,8,9,10,11,12,13,14,15);
      }
      #pragma unroll
      for (int j = 0; j < 4; ++j) {
        int nrow = wn * 64 + j * 16 + lm;
        // lane 0-15: K 0-7 & 16-23 ; lane 16-31: K 8-15 & 24-31
        const ushort_t* bbase = &ldsB[buf][nrow * KCH + s * 32];
        v8bf blo = *(const v8bf*)(bbase + half * 8);
        v8bf bhi = *(const v8bf*)(bbase + half * 8 + 16);
        bfrag[j] = __builtin_shufflevector(blo, bhi,
                     0,1,2,3,4,5,6,7,8,9,10,11,12,13,14,15);
      }
      #pragma unroll
      for (int i = 0; i < 2; ++i)
        #pragma unroll
        for (int j = 0; j < 4; ++j)
          acc[i][j] = __builtin_amdgcn_wmma_f32_16x16x32_bf16(
              false, afrag[i], false, bfrag[j], (short)0, acc[i][j], false, false);
    }
  };

  // ---- software pipeline, one barrier per K chunk ---------------------------
  issueA(0, 0);
  gatherB(0);
  commitA(0);
  commitB(0);
  __syncthreads();

  int cur = 0;
  for (int kk = 0; kk < KDIM; kk += KCH) {
    const int nk = kk + KCH;
    const bool more = nk < KDIM;                        // uniform branch
    if (more) { issueA(nk, cur ^ 1); gatherB(nk); }
    compute(cur);
    if (more) { commitA(cur ^ 1); commitB(cur ^ 1); }
    __syncthreads();
    cur ^= 1;
  }

  // ---- epilogue: C/D layout lane L -> N = L%16, M = vgpr + 8*(L/16) ---------
  #pragma unroll
  for (int i = 0; i < 2; ++i) {
    #pragma unroll
    for (int j = 0; j < 4; ++j) {
      const int pp = n0 + wn * 64 + j * 16 + lm;
      #pragma unroll
      for (int v = 0; v < 8; ++v) {
        const int o = m0 + wm * 32 + i * 16 + half * 8 + v;
        const size_t idx = ((size_t)(b * 256 + o) << 12) + pp;
        float val = acc[i][j][v];
        if (STAGE == 1) {
          float inv = g2[o] * rsqrtf(vr2[o] + BN_EPS);
          float y = fmaxf(fmaf(val, inv, fmaf(-mn2[o], inv, be2[o])), 0.f);
          actout[idx] = f2bf(y);
        } else {
          out[idx] = val + xres[idx];
        }
      }
    }
  }
}

// ---------------------------------------------------------------------------
extern "C" void kernel_launch(void* const* d_in, const int* in_sizes, int n_in,
                              void* d_out, int out_size, void* d_ws, size_t ws_size,
                              hipStream_t stream) {
  (void)in_sizes; (void)n_in; (void)out_size; (void)ws_size;
  const float* x   = (const float*)d_in[0];
  const float* hin = (const float*)d_in[1];
  const float* g1  = (const float*)d_in[2];
  const float* be1 = (const float*)d_in[3];
  const float* mn1 = (const float*)d_in[4];
  const float* vr1 = (const float*)d_in[5];
  const float* w1  = (const float*)d_in[6];
  const float* b1  = (const float*)d_in[7];
  const float* g2  = (const float*)d_in[8];
  const float* be2 = (const float*)d_in[9];
  const float* mn2 = (const float*)d_in[10];
  const float* vr2 = (const float*)d_in[11];
  const float* w2  = (const float*)d_in[12];
  const float* b2  = (const float*)d_in[13];
  float* out = (float*)d_out;

  ushort_t* A1 = (ushort_t*)d_ws;                       // 64 MB bf16 acts
  ushort_t* A2 = A1 + (size_t)32 * 256 * 4096;          // 64 MB bf16 acts
  ushort_t* Wb = A2 + (size_t)32 * 256 * 4096;          // 37.75 MB bf16 weights (reused)

  prep_bn_relu<<<32768, 256, 0, stream>>>(x, g1, be1, mn1, vr1, A1);

  dim3 grid(HW / 128, 256 / 128, 32);                   // (N tiles, M tiles, B)

  prep_weights<<<36864, 256, 0, stream>>>(w1, b1, hin, (unsigned*)Wb);
  hyperconv_wmma<1><<<grid, 256, 0, stream>>>(
      A1, Wb, g2, be2, mn2, vr2, nullptr, nullptr, A2);

  prep_weights<<<36864, 256, 0, stream>>>(w2, b2, hin, (unsigned*)Wb);
  hyperconv_wmma<2><<<grid, 256, 0, stream>>>(
      A2, Wb, nullptr, nullptr, nullptr, nullptr, x, out, nullptr);
}
